// TransformCTRNN_45896020525282
// MI455X (gfx1250) — matrix operand order repaired and checked by reading, not compiled, not run on hardware
//
#include <hip/hip_runtime.h>
#include <hip/hip_bf16.h>
#include <stdint.h>
#include <stddef.h>

// Problem dims (match reference)
#define T_STEPS 256
#define BATCH   64
#define IN_DIM  128
#define HID     256
#define ALPHA_F 0.1f
#define TIC_F   0.1f

typedef __attribute__((ext_vector_type(16))) __bf16 bf16x16;
typedef __attribute__((ext_vector_type(8)))  float  f32x8;
typedef __attribute__((ext_vector_type(4)))  float  f32x4;

union OperandU { f32x4 q[2]; bf16x16 m; };

__device__ __forceinline__ unsigned short f2bf(float f) {
  unsigned int u = __float_as_uint(f);
  u += 0x7FFFu + ((u >> 16) & 1u);          // round-to-nearest-even
  return (unsigned short)(u >> 16);
}

// A fragment (16x32 bf16, MxK): lane L holds row M=L&15; half=L>>4.
// elems 0..7  = K = half*8 + 0..7      (16 contiguous bytes)
// elems 8..15 = K = 16 + half*8 + 0..7 (16 contiguous bytes, +32B)
// p must already point at  row_base + half*8  (bf16 elements)
__device__ __forceinline__ bf16x16 load_a_frag(const unsigned short* p) {
  OperandU u;
  u.q[0] = *(const f32x4*)(p);
  u.q[1] = *(const f32x4*)(p + 16);
  return u.m;
}
// B fragment (32x16 bf16, KxN): lane L holds col N=L&15; half=L>>4.
// elems 0..7  = K = half*8 + 0..7      (16 contiguous bytes)
// elems 8..15 = K = 16 + half*8 + 0..7 (16 contiguous bytes, +32B)
// p must already point at  col_row_base + half*8
__device__ __forceinline__ bf16x16 load_b_frag(const unsigned short* p) {
  OperandU u;
  u.q[0] = *(const f32x4*)(p);
  u.q[1] = *(const f32x4*)(p + 16);
  return u.m;
}

__device__ __forceinline__ f32x8 wmma_bf16(bf16x16 a, bf16x16 b, f32x8 c) {
  // (neg_a, A, neg_b, B, c_mod, C, reuse_a, reuse_b)
  return __builtin_amdgcn_wmma_f32_16x16x32_bf16(false, a, false, b, (short)0, c,
                                                 false, false);
}

// ---------------------------------------------------------------- converts
__global__ void k_f32_to_bf16(const float* __restrict__ src,
                              unsigned short* __restrict__ dst, int n) {
  for (int i = blockIdx.x * blockDim.x + threadIdx.x; i < n;
       i += gridDim.x * blockDim.x)
    dst[i] = f2bf(src[i]);
}

// Th state layout (transposed for vector RMW): Th[(i*BATCH + b)*HID + j]
// Identity init over (i,j):  i = idx>>14 (idx/(B*H)),  j = idx & 255
__global__ void k_init_th(float* __restrict__ Th, int n) {
  for (int idx = blockIdx.x * blockDim.x + threadIdx.x; idx < n;
       idx += gridDim.x * blockDim.x)
    Th[idx] = ((idx >> 14) == (idx & (HID - 1))) ? 1.0f : 0.0f;
}

__global__ void k_init_h(float* __restrict__ hf, unsigned short* __restrict__ hb,
                         int n) {
  for (int i = blockIdx.x * blockDim.x + threadIdx.x; i < n;
       i += gridDim.x * blockDim.x) {
    hf[i] = 0.0f;
    hb[i] = 0;
  }
}

// ---------------------------------------------------- IP = input @ W_in^T + b
// A = input bf16 [T*B, I] row-major ; B = W_in bf16 [H, I] row-major (so
// B_mat[k][n] = W_in[n][k], contiguous in k). M=T*B=16384, N=HID, K=IN_DIM.
// 4 waves / block, 1 16x16 tile per wave, K=128 -> 4 WMMA steps.
__global__ void k_ip_gemm(const unsigned short* __restrict__ inb,
                          const unsigned short* __restrict__ winb,
                          const float* __restrict__ b_in,
                          float* __restrict__ ipf,
                          unsigned short* __restrict__ ipb) {
  const int wave = threadIdx.x >> 5;
  const int lane = threadIdx.x & 31;
  const int half = lane >> 4;
  const int lcol = lane & 15;

  const int tileId = blockIdx.x * 4 + wave;         // 16384 tiles
  const int mtile  = tileId >> 4;                   // 0..1023
  const int ntile  = tileId & 15;                   // 0..15

  const int arow = mtile * 16 + lcol;               // global row (t*B + b)
  const int bcol = ntile * 16 + lcol;               // hidden index n

  const unsigned short* ap = inb  + (size_t)arow * IN_DIM + half * 8;
  const unsigned short* bp = winb + (size_t)bcol * IN_DIM + half * 8;

  f32x8 acc = {};
#pragma unroll
  for (int ks = 0; ks < IN_DIM / 32; ++ks) {
    bf16x16 a = load_a_frag(ap + ks * 32);
    bf16x16 b = load_b_frag(bp + ks * 32);
    acc = wmma_bf16(a, b, acc);
  }

  const int   ccol = ntile * 16 + lcol;
  const float bias = b_in[ccol];
#pragma unroll
  for (int v = 0; v < 8; ++v) {
    const int    row = mtile * 16 + v + 8 * half;
    const size_t idx = (size_t)row * HID + ccol;
    const float  val = acc[v] + bias;
    ipf[idx] = val;
    ipb[idx] = f2bf(val);
  }
}

// ------------------------------------------- fused tm / Th-update / ti kernel
// One block per output row i (grid = HID). Computes, for all (j, b):
//   tm[i,j,b]  = sum_k transform[i*H+j, k] * ip_t[b, k]        (WMMA, bf16)
//   Th[i,b,j]  = 0.9*Th + 0.1*tm                               (fp32, in-place)
//   ti[i,b]   += Th[i,b,j] * ip_t[b, j]                        (fused reduce)
// 8 waves; wave w owns j-tiles {2w, 2w+1}; 4 b-tiles held concurrently so the
// A fragment (transform row) is reused 4x per load. Th stored [i][b][j] so the
// per-lane RMW is two b128 loads + two b128 stores (j contiguous).
__global__ void __launch_bounds__(256)
k_step(const unsigned short* __restrict__ tfb,   // transform bf16 [H*H, H]
       const unsigned short* __restrict__ ipb,   // ip bf16 [T, B, H]
       const float* __restrict__ ipf,            // ip fp32 [T, B, H]
       float* __restrict__ Th,                   // [H, B, H]  (i, b, j)
       float* __restrict__ ti,                   // [H, B]
       int t) {
  __shared__ __align__(16) unsigned short s_ip[BATCH * HID];  // 32 KB
  __shared__ float s_ti[BATCH];

  const int tid  = threadIdx.x;
  const int i    = blockIdx.x;                   // output row 0..255
  const int wave = tid >> 5;
  const int lane = tid & 31;
  const int half = lane >> 4;
  const int lcol = lane & 15;

  // Stage ip_t (bf16, [B][H]) into LDS: 2048 x 16B
  {
    const f32x4* src = (const f32x4*)(ipb + (size_t)t * BATCH * HID);
    f32x4*       dst = (f32x4*)s_ip;
    for (int k = tid; k < (BATCH * HID) / 8; k += 256) dst[k] = src[k];
  }
  if (tid < BATCH) s_ti[tid] = 0.0f;
  __syncthreads();

#pragma unroll
  for (int jj = 0; jj < 2; ++jj) {
    const int jt = wave * 2 + jj;                // j-tile 0..15
    // A row: transform row r = i*256 + jt*16 + lcol
    const unsigned short* arow =
        tfb + ((size_t)(i * HID + jt * 16 + lcol)) * HID + half * 8;
    __builtin_prefetch(arow, 0, 1);              // global_prefetch_b8

    f32x8 acc[4] = {};
#pragma unroll
    for (int ks = 0; ks < HID / 32; ++ks) {
      bf16x16 a = load_a_frag(arow + ks * 32);
#pragma unroll
      for (int bt = 0; bt < 4; ++bt) {
        const unsigned short* bp =
            s_ip + (size_t)(bt * 16 + lcol) * HID + ks * 32 + half * 8;
        acc[bt] = wmma_bf16(a, load_b_frag(bp), acc[bt]);
      }
    }

    // Fused epilogue: vectorized Th RMW + partial ti
    const int jbase = jt * 16 + 8 * half;        // 8 contiguous j per lane
#pragma unroll
    for (int bt = 0; bt < 4; ++bt) {
      const int b = bt * 16 + lcol;
      // ip_t[b, jbase..jbase+7] fp32 (contiguous, 32B)
      f32x4 ip0, ip1;
      {
        const float* p = ipf + ((size_t)t * BATCH + b) * HID + jbase;
        ip0 = ((const f32x4*)p)[0];
        ip1 = ((const f32x4*)p)[1];
      }
      float* thp = Th + ((size_t)(i * BATCH + b)) * HID + jbase;
      f32x4  th0 = ((const f32x4*)thp)[0];
      f32x4  th1 = ((const f32x4*)thp)[1];
      float  partial = 0.0f;
#pragma unroll
      for (int v = 0; v < 4; ++v) {
        th0[v] = th0[v] * (1.0f - TIC_F) + acc[bt][v] * TIC_F;
        partial += th0[v] * ip0[v];
      }
#pragma unroll
      for (int v = 0; v < 4; ++v) {
        th1[v] = th1[v] * (1.0f - TIC_F) + acc[bt][v + 4] * TIC_F;
        partial += th1[v] * ip1[v];
      }
      ((f32x4*)thp)[0] = th0;
      ((f32x4*)thp)[1] = th1;

      partial += __shfl_xor(partial, 16);        // combine both 8-row halves
      if (lane < 16) atomicAdd(&s_ti[b], partial);  // ds_add_f32
    }
  }
  __syncthreads();
  if (tid < BATCH) ti[(size_t)i * BATCH + tid] = s_ti[tid];
}

// ------------------------------------------------- hidden-state update kernel
// h_pre = h @ W_h^T (WMMA, M=B, N=HID, K=HID); h_new = relu(ti + h_pre + b_h);
// h' = 0.9*h + 0.1*h_new; writes output[t] and next-step h buffers.
__global__ void __launch_bounds__(128)
k_hstep(const unsigned short* __restrict__ hb_in,   // h bf16 [B, H]
        const unsigned short* __restrict__ whb,     // W_h bf16 [H, H]
        const float* __restrict__ ti,               // [H, B]
        const float* __restrict__ b_h,              // [H]
        const float* __restrict__ h_in,             // h fp32 [B, H]
        float* __restrict__ h_out,
        unsigned short* __restrict__ hb_out,
        float* __restrict__ out,                    // [T, B, H]
        int t) {
  const int wave = threadIdx.x >> 5;
  const int lane = threadIdx.x & 31;
  const int half = lane >> 4;
  const int lcol = lane & 15;

  const int tileId = blockIdx.x * 4 + wave;   // 64 tiles
  const int mtile  = tileId >> 4;             // b tile 0..3
  const int ntile  = tileId & 15;             // h tile 0..15

  const int arow = mtile * 16 + lcol;         // batch index for A
  const int bcol = ntile * 16 + lcol;         // hidden index for B/C

  const unsigned short* ap = hb_in + (size_t)arow * HID + half * 8;
  const unsigned short* bp = whb   + (size_t)bcol * HID + half * 8;

  f32x8 acc = {};
#pragma unroll
  for (int ks = 0; ks < HID / 32; ++ks)
    acc = wmma_bf16(load_a_frag(ap + ks * 32), load_b_frag(bp + ks * 32), acc);

  // ti[bcol, bbase..bbase+7] contiguous fp32
  const int bbase = mtile * 16 + 8 * half;
  float tiv[8];
  {
    const float* p = ti + (size_t)bcol * BATCH + bbase;
    ((f32x4*)tiv)[0] = ((const f32x4*)p)[0];
    ((f32x4*)tiv)[1] = ((const f32x4*)p)[1];
  }
  const float bias = b_h[bcol];

#pragma unroll
  for (int v = 0; v < 8; ++v) {
    const int    b   = bbase + v;
    const size_t hix = (size_t)b * HID + bcol;
    const float  pre = acc[v] + tiv[v] + bias;
    const float  hn  = fmaxf(pre, 0.0f);
    const float  hu  = h_in[hix] * (1.0f - ALPHA_F) + ALPHA_F * hn;
    h_out[hix]  = hu;
    hb_out[hix] = f2bf(hu);
    out[((size_t)t * BATCH + b) * HID + bcol] = hu;
  }
}

// ------------------------------------------------------------------- launch
extern "C" void kernel_launch(void* const* d_in, const int* in_sizes, int n_in,
                              void* d_out, int out_size, void* d_ws,
                              size_t ws_size, hipStream_t stream) {
  const float* input     = (const float*)d_in[0];   // [T, B, I]
  const float* W_in      = (const float*)d_in[1];   // [H, I]
  const float* b_in      = (const float*)d_in[2];   // [H]
  const float* W_h       = (const float*)d_in[3];   // [H, H]
  const float* b_h       = (const float*)d_in[4];   // [H]
  const float* transform = (const float*)d_in[5];   // [H*H, H]
  float* out = (float*)d_out;

  char*  ws  = (char*)d_ws;
  size_t off = 0;
  auto carve = [&](size_t bytes) -> char* {
    char* p = ws + off;
    off = (off + bytes + 255) & ~(size_t)255;
    return p;
  };
  unsigned short* tfb  = (unsigned short*)carve((size_t)HID * HID * HID * 2); // 33.5MB
  unsigned short* whb  = (unsigned short*)carve((size_t)HID * HID * 2);
  unsigned short* inb  = (unsigned short*)carve((size_t)T_STEPS * BATCH * IN_DIM * 2);
  unsigned short* winb = (unsigned short*)carve((size_t)HID * IN_DIM * 2);
  float*          ipf  = (float*)carve((size_t)T_STEPS * BATCH * HID * 4);    // 16.8MB
  unsigned short* ipb  = (unsigned short*)carve((size_t)T_STEPS * BATCH * HID * 2);
  float*          Th   = (float*)carve((size_t)HID * BATCH * HID * 4);        // 16.8MB
  float*          ti   = (float*)carve((size_t)HID * BATCH * 4);
  float*          hf[2];
  unsigned short* hb[2];
  hf[0] = (float*)carve((size_t)BATCH * HID * 4);
  hf[1] = (float*)carve((size_t)BATCH * HID * 4);
  hb[0] = (unsigned short*)carve((size_t)BATCH * HID * 2);
  hb[1] = (unsigned short*)carve((size_t)BATCH * HID * 2);

  // One-time (per call) precision lowering to bf16
  {
    int n = HID * HID * HID;
    k_f32_to_bf16<<<(n + 255) / 256, 256, 0, stream>>>(transform, tfb, n);
    n = HID * HID;
    k_f32_to_bf16<<<(n + 255) / 256, 256, 0, stream>>>(W_h, whb, n);
    n = T_STEPS * BATCH * IN_DIM;
    k_f32_to_bf16<<<(n + 255) / 256, 256, 0, stream>>>(input, inb, n);
    n = HID * IN_DIM;
    k_f32_to_bf16<<<(n + 255) / 256, 256, 0, stream>>>(W_in, winb, n);
  }
  // State init
  {
    int n = HID * BATCH * HID;
    k_init_th<<<(n + 255) / 256, 256, 0, stream>>>(Th, n);
    n = BATCH * HID;
    k_init_h<<<(n + 255) / 256, 256, 0, stream>>>(hf[0], hb[0], n);
  }
  // Hoisted input projection for all timesteps: [16384,128] x [128,256]
  k_ip_gemm<<<(T_STEPS * BATCH / 16) * (HID / 16) / 4, 128, 0, stream>>>(
      inb, winb, b_in, ipf, ipb);

  // Sequential recurrence
  for (int t = 0; t < T_STEPS; ++t) {
    k_step<<<HID, 256, 0, stream>>>(tfb, ipb, ipf, Th, ti, t);
    const int pi = t & 1, po = (t + 1) & 1;
    k_hstep<<<(BATCH / 16) * (HID / 16) / 4, 128, 0, stream>>>(
        hb[pi], whb, ti, b_h, hf[pi], hf[po], hb[po], out, t);
  }

  // h_final (second tuple element) after output: T is even -> final h in buf 0
  hipMemcpyAsync(out + (size_t)T_STEPS * BATCH * HID, hf[(T_STEPS) & 1],
                 (size_t)BATCH * HID * sizeof(float), hipMemcpyDeviceToDevice,
                 stream);
}
